// RelativeMultiHeadAttention_26843545600459
// MI455X (gfx1250) — matrix operand order repaired and checked by reading, not compiled, not run on hardware
//
#include <hip/hip_runtime.h>

// ---------------------------------------------------------------------------
// Transformer-XL relative multi-head attention for MI455X (gfx1250).
// All heavy math runs through v_wmma_f32_16x16x32_bf16 (wave32 WMMA).
// B=8, T=1024, H=8, d=64, D_MODEL=QKV=OUT=512, W=2*T-1=2047.
// ---------------------------------------------------------------------------

typedef unsigned short u16;
typedef unsigned int   u32;
typedef __attribute__((ext_vector_type(16))) __bf16 v16bf;
typedef __attribute__((ext_vector_type(8)))  float  v8f;

#define T_SEQ   1024
#define NHEAD   8
#define HDIM    64
#define DMODEL  512
#define BATCH   8
#define WREL    2047

union FragB {            // one WMMA bf16 operand: 16x32 (A) or 32x16 (B)
    v16bf v;
    uint4 u[2];
};

static __device__ __forceinline__ u16 f2bf(float f) {
    u32 u = __float_as_uint(f);
    u32 r = (u + 0x7FFFu + ((u >> 16) & 1u)) >> 16;   // round-to-nearest-even
    return (u16)r;
}
static __device__ __forceinline__ u32 pk2(float a, float b) {
    return (u32)f2bf(a) | ((u32)f2bf(b) << 16);
}
static __device__ __forceinline__ v8f vzero() {
    v8f z;
#pragma unroll
    for (int i = 0; i < 8; ++i) z[i] = 0.0f;
    return z;
}
static __device__ __forceinline__ v8f wmma_bf16(const FragB& a, const FragB& b, v8f c) {
    return __builtin_amdgcn_wmma_f32_16x16x32_bf16(
        /*neg_a=*/false, a.v, /*neg_b=*/false, b.v,
        /*c_mod=*/(short)0, c, /*reuse_a=*/false, /*reuse_b=*/false);
}

// ---------------------------------------------------------------------------
// Generic bf16 WMMA GEMM:  C[M,512] = A[M,512] @ B[512,512] (+bias)
// A source is fp32 or bf16 (converted while staging to LDS); B (weights)
// is always fp32 and is transposed into LDS for contiguous K-fragments.
// Block = 256 threads (8 waves); tile = 128x128x64; 128 WMMA / wave / launch.
// ---------------------------------------------------------------------------
template <bool A_BF16, bool OUT_BF16, bool HAS_BIAS>
__global__ __launch_bounds__(256) void gemm_wmma_bf16(
    const void* __restrict__ Ap, const float* __restrict__ Bw,
    const float* __restrict__ bias, void* __restrict__ Cp, int M)
{
    constexpr int BM = 128, BN = 128, BK = 64, LDT = 72;  // 72*2B rows: 16B aligned
    __shared__ u16 sA[BM][LDT];     // A tile, row-major [m][k], bf16
    __shared__ u16 sBT[BN][LDT];    // B tile, transposed [n][k], bf16

    const int tid  = threadIdx.x;
    const int lane = tid & 31;
    const int w    = tid >> 5;
    const int M0   = blockIdx.x * BM;
    const int N0   = blockIdx.y * BN;
    const int wm0  = (w & 1) * 64;
    const int wn0  = (w >> 1) * 32;
    const int lhalf = (lane >> 4) & 1;
    const int l15   = lane & 15;

    v8f acc[4][2];
#pragma unroll
    for (int i = 0; i < 4; ++i)
#pragma unroll
        for (int j = 0; j < 2; ++j) acc[i][j] = vzero();

    for (int k0 = 0; k0 < 512; k0 += BK) {
        __syncthreads();
        // ---- stage A tile (convert fp32 -> bf16 if needed) ----
        if (A_BF16) {
            const u16* Ab = (const u16*)Ap;
            for (int i = tid; i < BM * 8; i += 256) {
                int r = i >> 3, c8 = (i & 7) * 8;
                int gm = M0 + r;
                uint4 val; val.x = val.y = val.z = val.w = 0u;
                if (gm < M) val = *(const uint4*)(Ab + (size_t)gm * 512 + k0 + c8);
                *(uint4*)&sA[r][c8] = val;
            }
        } else {
            const float* Af = (const float*)Ap;
            for (int i = tid; i < BM * 16; i += 256) {
                int r = i >> 4, c4 = (i & 15) * 4;
                int gm = M0 + r;
                uint2 val; val.x = val.y = 0u;
                if (gm < M) {
                    float4 f = *(const float4*)(Af + (size_t)gm * 512 + k0 + c4);
                    val.x = pk2(f.x, f.y);
                    val.y = pk2(f.z, f.w);
                }
                *(uint2*)&sA[r][c4] = val;
            }
        }
        // ---- stage B tile transposed: sBT[n][k] = W[k0+k][N0+n] ----
        for (int i = tid; i < BK * 32; i += 256) {
            int kr = i >> 5, n4 = (i & 31) * 4;
            float4 wv = *(const float4*)(Bw + (size_t)(k0 + kr) * 512 + N0 + n4);
            sBT[n4 + 0][kr] = f2bf(wv.x);
            sBT[n4 + 1][kr] = f2bf(wv.y);
            sBT[n4 + 2][kr] = f2bf(wv.z);
            sBT[n4 + 3][kr] = f2bf(wv.w);
        }
        __syncthreads();

        // ---- compute: 2 K-substeps x (2 B-frags, 4 A-frags, 8 WMMA) ----
#pragma unroll
        for (int kk = 0; kk < BK; kk += 32) {
            FragB bF[2];
#pragma unroll
            for (int nt = 0; nt < 2; ++nt) {
                int n  = wn0 + nt * 16 + l15;
                int kb = kk + lhalf * 16;          // B: lanes 0-15 K=0-15, 16-31 K=16-31
                bF[nt].u[0] = *(const uint4*)&sBT[n][kb];
                bF[nt].u[1] = *(const uint4*)&sBT[n][kb + 8];
            }
#pragma unroll
            for (int mt = 0; mt < 4; ++mt) {
                FragB aF;
                int m  = wm0 + mt * 16 + l15;
                int kb = kk + lhalf * 16;          // A: lanes 0-15 K=0-15, 16-31 K=16-31
                aF.u[0] = *(const uint4*)&sA[m][kb];
                aF.u[1] = *(const uint4*)&sA[m][kb + 8];
#pragma unroll
                for (int nt = 0; nt < 2; ++nt)
                    acc[mt][nt] = wmma_bf16(aF, bF[nt], acc[mt][nt]);
            }
        }
    }

    // ---- epilogue: C layout VGPR r -> row r + 8*half, col = lane&15 ----
#pragma unroll
    for (int mt = 0; mt < 4; ++mt)
#pragma unroll
        for (int nt = 0; nt < 2; ++nt)
#pragma unroll
            for (int r = 0; r < 8; ++r) {
                int m = M0 + wm0 + mt * 16 + r + lhalf * 8;
                int n = N0 + wn0 + nt * 16 + l15;
                if (m < M) {
                    float val = acc[mt][nt][r];
                    if (HAS_BIAS) val += bias[n];
                    if (OUT_BF16)
                        ((u16*)Cp)[(size_t)m * 512 + n] = f2bf(val);
                    else
                        ((float*)Cp)[(size_t)m * 512 + n] = val;
                }
            }
}

// ---------------------------------------------------------------------------
// Flash attention with relative position (online softmax, never materializes
// T x T).  One wave = 16 query rows of one (b,h); 8 waves / block share the
// K and V tiles in LDS (K row-major == natural B-fragment layout; V
// transposed).  Pos term uses the "skew" trick: 16x32 WMMA panel against 32
// consecutive p rows, spilled to LDS, gathered on the diagonal j=n-m+15.
// score[t,s] = ((q+u)[t].k[s] + (q+v)[t].p[T-1+s-t]) / 8
// ---------------------------------------------------------------------------
__global__ __launch_bounds__(256) void flash_relattn(
    const float* __restrict__ q,  const u16* __restrict__ k,
    const u16* __restrict__ v,    const u16* __restrict__ p,
    const float* __restrict__ uB, const float* __restrict__ vB,
    u16* __restrict__ outA)
{
    __shared__ u16   sK[32][72];         // K tile row-major [s][c]   (4.6 KB)
    __shared__ u16   sVT[64][40];        // V tile transposed [c][s]  (5.1 KB)
    __shared__ float panelS[8][16][33];  // per-wave pos panel f32    (16.9 KB)
    __shared__ u16   sP[8][16][40];      // per-wave probs bf16       (10.2 KB)

    const int tid  = threadIdx.x;
    const int lane = tid & 31;
    const int w    = tid >> 5;                   // 0..7
    const int bid  = blockIdx.x;                 // b*64 + h*8 + tb
    const int tb   = bid & 7;
    const int h    = (bid >> 3) & 7;
    const int b    = bid >> 6;
    const int t0   = tb * 128 + w * 16;
    const int lhalf = (lane >> 4) & 1;
    const int l15   = lane & 15;

    // ---- build A-fragments for (q+u) and (q+v), fp32 adds then bf16 ----
    FragB quF[2], qvF[2];
    {
        const int    mrow = t0 + l15;
        const float* qrow = q + ((size_t)(b * T_SEQ + mrow)) * 512 + h * HDIM;
        const float* uh   = uB + h * HDIM;
        const float* vh   = vB + h * HDIM;
        const int    kb   = lhalf * 16;
#pragma unroll
        for (int f = 0; f < 2; ++f) {
            int c0 = f * 32 + kb;
#pragma unroll
            for (int g = 0; g < 2; ++g) {
                int cg = c0 + g * 8;
                u32 tu[4], tv[4];
#pragma unroll
                for (int d = 0; d < 4; ++d) {
                    int c = cg + 2 * d;
                    float a0 = qrow[c], a1 = qrow[c + 1];
                    tu[d] = pk2(a0 + uh[c], a1 + uh[c + 1]);
                    tv[d] = pk2(a0 + vh[c], a1 + vh[c + 1]);
                }
                uint4 U; U.x = tu[0]; U.y = tu[1]; U.z = tu[2]; U.w = tu[3];
                uint4 V; V.x = tv[0]; V.y = tv[1]; V.z = tv[2]; V.w = tv[3];
                quF[f].u[g] = U;
                qvF[f].u[g] = V;
            }
        }
    }

    v8f acc[4];
#pragma unroll
    for (int nt = 0; nt < 4; ++nt) acc[nt] = vzero();
    float rowM[8], rowL[8];
#pragma unroll
    for (int r = 0; r < 8; ++r) { rowM[r] = -1e30f; rowL[r] = 0.0f; }

    for (int s0 = 0; s0 < T_SEQ; s0 += 32) {
        __syncthreads();
        // ---- cooperative V transpose into LDS: sVT[c][s] ----
        for (int i = tid; i < 32 * 32; i += 256) {
            int s = i >> 5, c2 = (i & 31) * 2;
            u32 val = *(const u32*)(v + ((size_t)(b * T_SEQ + s0 + s)) * 512 + h * HDIM + c2);
            sVT[c2][s]     = (u16)(val & 0xFFFFu);
            sVT[c2 + 1][s] = (u16)(val >> 16);
        }
        // ---- cooperative K tile stage (row-major, one b128 per thread) ----
        {
            int r = tid >> 3, c8 = (tid & 7) * 8;
            uint4 kv = *(const uint4*)(k + ((size_t)(b * T_SEQ + s0 + r)) * 512 + h * HDIM + c8);
            *(uint4*)&sK[r][c8] = kv;
        }
        __syncthreads();

        // ---- prefetch next K/V tiles into caches (global_prefetch_b8) ----
        if (s0 + 32 < T_SEQ) {
            size_t nrow = (size_t)(b * T_SEQ + s0 + 32 + lane) * 512 + h * HDIM;
            __builtin_prefetch((const void*)(k + nrow), 0, 1);
            __builtin_prefetch((const void*)(v + nrow), 0, 1);
        }

        float sc[2][8];
#pragma unroll
        for (int sub = 0; sub < 2; ++sub) {
            const int s16  = s0 + sub * 16;
            const int koff = lhalf * 16;

            // content scores: (q+u) @ k^T, B-frags from the shared LDS K tile
            v8f cacc = vzero();
            const u16* krow = &sK[sub * 16 + l15][koff];
#pragma unroll
            for (int f = 0; f < 2; ++f) {
                FragB kF;
                kF.u[0] = *(const uint4*)(krow + f * 32);
                kF.u[1] = *(const uint4*)(krow + f * 32 + 8);
                cacc = wmma_bf16(quF[f], kF, cacc);
            }

            // pos panel: (q+v) @ p[base .. base+31]^T  -> LDS
            const int base = (T_SEQ - 1) + s16 - t0 - 15;
#pragma unroll
            for (int jb = 0; jb < 2; ++jb) {
                v8f pacc = vzero();
                const u16* prow = p + ((size_t)(base + jb * 16 + l15)) * 512 + h * HDIM + koff;
#pragma unroll
                for (int f = 0; f < 2; ++f) {
                    FragB pF;
                    pF.u[0] = *(const uint4*)(prow + f * 32);
                    pF.u[1] = *(const uint4*)(prow + f * 32 + 8);
                    pacc = wmma_bf16(qvF[f], pF, pacc);
                }
#pragma unroll
                for (int r = 0; r < 8; ++r)
                    panelS[w][lhalf * 8 + r][jb * 16 + l15] = pacc[r];
            }
            // diagonal gather (same-wave LDS ops are in-order, no barrier)
#pragma unroll
            for (int r = 0; r < 8; ++r) {
                int m = lhalf * 8 + r;
                int j = l15 - m + 15;                 // in [0,30]
                sc[sub][r] = (cacc[r] + panelS[w][m][j]) * 0.125f;
            }
        }

        // ---- online softmax over this 32-key block ----
#pragma unroll
        for (int r = 0; r < 8; ++r) {
            float mx = fmaxf(sc[0][r], sc[1][r]);
#pragma unroll
            for (int off = 1; off < 16; off <<= 1)
                mx = fmaxf(mx, __shfl_xor(mx, off, 16));
            float mnew = fmaxf(rowM[r], mx);
            float corr = __expf(rowM[r] - mnew);
            rowM[r] = mnew;
            float p0 = __expf(sc[0][r] - mnew);
            float p1 = __expf(sc[1][r] - mnew);
            float rs = p0 + p1;
#pragma unroll
            for (int off = 1; off < 16; off <<= 1)
                rs += __shfl_xor(rs, off, 16);
            rowL[r] = rowL[r] * corr + rs;
#pragma unroll
            for (int nt = 0; nt < 4; ++nt) acc[nt] *= corr;
            int m = lhalf * 8 + r;
            sP[w][m][l15]      = f2bf(p0);
            sP[w][m][16 + l15] = f2bf(p1);
        }

        // ---- O += P @ V  (A-frag from sP, B-frags from sVT) ----
        FragB pFrag;
        {
            int m  = l15;
            int kb = lhalf * 16;
            pFrag.u[0] = *(const uint4*)&sP[w][m][kb];
            pFrag.u[1] = *(const uint4*)&sP[w][m][kb + 8];
        }
#pragma unroll
        for (int nt = 0; nt < 4; ++nt) {
            FragB vF;
            int c     = nt * 16 + l15;
            int koff2 = lhalf * 16;
            vF.u[0] = *(const uint4*)&sVT[c][koff2];
            vF.u[1] = *(const uint4*)&sVT[c][koff2 + 8];
            acc[nt] = wmma_bf16(pFrag, vF, acc[nt]);
        }
    }

    // ---- finalize: O / l, bf16 store to (b, t, h*64 + c) ----
#pragma unroll
    for (int r = 0; r < 8; ++r) {
        float inv = 1.0f / rowL[r];
        int m = t0 + lhalf * 8 + r;
#pragma unroll
        for (int nt = 0; nt < 4; ++nt)
            outA[((size_t)(b * T_SEQ + m)) * 512 + h * HDIM + nt * 16 + l15] =
                f2bf(acc[nt][r] * inv);
    }
}

// ---------------------------------------------------------------------------
// Host-side orchestration
// Inputs: x, pos_emb, Wq, bq, Wk, Wv, Wp, Wo, bo, u, v
// ---------------------------------------------------------------------------
extern "C" void kernel_launch(void* const* d_in, const int* in_sizes, int n_in,
                              void* d_out, int out_size, void* d_ws, size_t ws_size,
                              hipStream_t stream)
{
    (void)in_sizes; (void)n_in; (void)out_size; (void)ws_size;

    const float* x       = (const float*)d_in[0];
    const float* pos_emb = (const float*)d_in[1];
    const float* Wq      = (const float*)d_in[2];
    const float* bq      = (const float*)d_in[3];
    const float* Wk      = (const float*)d_in[4];
    const float* Wv      = (const float*)d_in[5];
    const float* Wp      = (const float*)d_in[6];
    const float* Wo      = (const float*)d_in[7];
    const float* bo      = (const float*)d_in[8];
    const float* uvec    = (const float*)d_in[9];
    const float* vvec    = (const float*)d_in[10];
    float* out           = (float*)d_out;

    const size_t MROWS = (size_t)BATCH * T_SEQ;          // 8192
    char* ws = (char*)d_ws;
    const size_t SZ_Q = MROWS * 512 * sizeof(float);     // 16 MiB
    const size_t SZ_K = MROWS * 512 * sizeof(u16);       //  8 MiB
    const size_t SZ_P = (size_t)2048 * 512 * sizeof(u16);//  2 MiB (padded row 2047)

    float* qbuf = (float*)ws;                 ws += SZ_Q;
    u16*   kbuf = (u16*)ws;                   ws += SZ_K;
    u16*   vbuf = (u16*)ws;                   ws += SZ_K;
    u16*   pbuf = (u16*)ws;                   ws += SZ_P;
    u16*   abuf = (u16*)ws;

    dim3 blk(256);
    dim3 gXW(64, 4);   // 8192/128 x 512/128
    dim3 gPW(16, 4);   // 2048/128 x 512/128

    // projections (fp32 in, bf16 WMMA core)
    gemm_wmma_bf16<false, false, true ><<<gXW, blk, 0, stream>>>(x,       Wq, bq,      qbuf, (int)MROWS);
    gemm_wmma_bf16<false, true,  false><<<gXW, blk, 0, stream>>>(x,       Wk, nullptr, kbuf, (int)MROWS);
    gemm_wmma_bf16<false, true,  false><<<gXW, blk, 0, stream>>>(x,       Wv, nullptr, vbuf, (int)MROWS);
    gemm_wmma_bf16<false, true,  false><<<gPW, blk, 0, stream>>>(pos_emb, Wp, nullptr, pbuf, WREL);

    // flash attention with relative position: 8*8*8 = 512 blocks, 8 waves each
    flash_relattn<<<dim3(BATCH * NHEAD * (T_SEQ / 128)), dim3(256), 0, stream>>>(
        qbuf, kbuf, vbuf, pbuf, uvec, vvec, abuf);

    // output projection (bf16 A, fp32 out + bias)
    gemm_wmma_bf16<true,  false, true ><<<gXW, blk, 0, stream>>>(abuf,    Wo, bo,      out,  (int)MROWS);
}
